// AdaptiveHierarchicalAttention_9045201125571
// MI455X (gfx1250) — matrix operand order repaired and checked by reading, not compiled, not run on hardware
//
#include <hip/hip_runtime.h>
#include <math.h>

// ---------------- CDNA5 WMMA types ----------------
typedef __attribute__((ext_vector_type(16))) __bf16 v16bf;
typedef __attribute__((ext_vector_type(8)))  float  v8f;

union V16 {
  v16bf v;
  unsigned short h[16];
  unsigned int   u[8];
};

__device__ __forceinline__ unsigned short f2bf(float f) {
  // round-to-nearest-even fp32 -> bf16
  unsigned int u = __float_as_uint(f);
  u += 0x7FFFu + ((u >> 16) & 1u);
  return (unsigned short)(u >> 16);
}

__device__ __forceinline__ v8f v8f_zero() {
  v8f z;
#pragma unroll
  for (int i = 0; i < 8; ++i) z[i] = 0.0f;
  return z;
}

__device__ __forceinline__ v8f wmma_bf16(const V16& a, const V16& b, v8f c) {
  // D = A(16x32 bf16) x B(32x16 bf16) + C(16x16 f32)
  return __builtin_amdgcn_wmma_f32_16x16x32_bf16(
      false, a.v, false, b.v, (short)0, c, false, false);
}

// =====================================================================
// GEMM: C[M,N] = A[M,K] @ W[N,K]^T + bias[N] (+ res[M,N] if RES)
// Output fp32 (C) or bf16 (Cb) selected at compile time.
// Block: 256 threads (8 waves); tile BM=64 x BN=128, K-step 32.
// Each wave computes a 32x32 C sub-tile = 2x2 WMMA tiles.
// =====================================================================
#define BM 64
#define BN 128
#define BK 32
#define LDP 36  // padded LDS row stride in ushorts (avoid bank conflicts)

template <bool RES, bool BF16OUT>
__global__ void __launch_bounds__(256)
gemm_bias_wmma(const float* __restrict__ A, long lda,
               const float* __restrict__ W, long ldw,
               const float* __restrict__ bias,
               const float* __restrict__ res,
               float* __restrict__ C,
               unsigned short* __restrict__ Cb,
               int M, int N, int K) {
  __shared__ unsigned short Alds[BM * LDP];
  __shared__ unsigned short Wlds[BN * LDP];

  const int tid  = threadIdx.x;
  const int lane = tid & 31;
  const int wave = tid >> 5;
  const int wm   = wave >> 2;  // 0..1  (M direction, 32 rows each)
  const int wn   = wave & 3;   // 0..3  (N direction, 32 cols each)
  const long row0 = (long)blockIdx.x * BM;
  const long col0 = (long)blockIdx.y * BN;

  v8f acc[2][2];
#pragma unroll
  for (int i = 0; i < 2; ++i)
#pragma unroll
    for (int j = 0; j < 2; ++j) acc[i][j] = v8f_zero();

  // staging assignments
  const int arow = tid >> 2;            // 0..63
  const int acol = (tid & 3) * 8;       // 0,8,16,24
  const int wrow = tid >> 1;            // 0..127
  const int wcol = (tid & 1) * 16;      // 0,16

  const int mlow  = lane & 15;
  const int aklow = (lane < 16) ? 0 : 8;   // A-operand K runs: {k..k+7, k+16..k+23}
  const int bklow = (lane < 16) ? 0 : 8;   // B-operand K runs: {k..k+7, k+16..k+23}

  for (int kk = 0; kk < K; kk += BK) {
    // ---- stage A tile (64x32 fp32 -> bf16 LDS) ----
    {
      const float* ap = A + (row0 + arow) * lda + kk + acol;
      float4 f0 = *(const float4*)ap;
      float4 f1 = *(const float4*)(ap + 4);
      unsigned short* d = &Alds[arow * LDP + acol];
      d[0] = f2bf(f0.x); d[1] = f2bf(f0.y); d[2] = f2bf(f0.z); d[3] = f2bf(f0.w);
      d[4] = f2bf(f1.x); d[5] = f2bf(f1.y); d[6] = f2bf(f1.z); d[7] = f2bf(f1.w);
      __builtin_prefetch((const void*)(ap + BK), 0, 1);  // global_prefetch_b8
    }
    // ---- stage W tile (128x32 fp32 -> bf16 LDS) ----
    {
      const float* wp = W + (col0 + wrow) * ldw + kk + wcol;
      unsigned short* d = &Wlds[wrow * LDP + wcol];
#pragma unroll
      for (int q = 0; q < 4; ++q) {
        float4 f = *(const float4*)(wp + 4 * q);
        d[4 * q + 0] = f2bf(f.x); d[4 * q + 1] = f2bf(f.y);
        d[4 * q + 2] = f2bf(f.z); d[4 * q + 3] = f2bf(f.w);
      }
      __builtin_prefetch((const void*)(wp + BK), 0, 1);
    }
    __syncthreads();

    // ---- build operands from LDS per the CDNA5 16-bit WMMA layouts ----
    V16 a[2], b[2];
#pragma unroll
    for (int sm = 0; sm < 2; ++sm) {
      const unsigned short* p = &Alds[(wm * 32 + sm * 16 + mlow) * LDP];
#pragma unroll
      for (int j = 0; j < 4; ++j) {
        a[sm].u[j]     = *(const unsigned int*)(p + aklow + 2 * j);
        a[sm].u[4 + j] = *(const unsigned int*)(p + aklow + 16 + 2 * j);
      }
    }
#pragma unroll
    for (int sn = 0; sn < 2; ++sn) {
      const unsigned short* p = &Wlds[(wn * 32 + sn * 16 + mlow) * LDP + bklow];
#pragma unroll
      for (int j = 0; j < 4; ++j) { b[sn].u[j] = *(const unsigned int*)(p + 2 * j); b[sn].u[4 + j] = *(const unsigned int*)(p + 16 + 2 * j); }
    }

#pragma unroll
    for (int sm = 0; sm < 2; ++sm)
#pragma unroll
      for (int sn = 0; sn < 2; ++sn)
        acc[sm][sn] = wmma_bf16(a[sm], b[sn], acc[sm][sn]);

    __syncthreads();
  }

  // ---- epilogue: bias (+ residual), fp32 or bf16 store ----
  const int vofs = (lane < 16) ? 0 : 8;
  const int cl   = lane & 15;
#pragma unroll
  for (int sm = 0; sm < 2; ++sm) {
    const long rowb = row0 + wm * 32 + sm * 16 + vofs;
#pragma unroll
    for (int sn = 0; sn < 2; ++sn) {
      const long col = col0 + wn * 32 + sn * 16 + cl;
      const float bv = bias[col];
      const long base = rowb * (long)N + col;
      float* __restrict__ cp = BF16OUT ? nullptr : (C + base);
      unsigned short* __restrict__ cb = BF16OUT ? (Cb + base) : nullptr;
      const float* __restrict__ rp = RES ? (res + base) : nullptr;
#pragma unroll
      for (int v = 0; v < 8; ++v) {
        float val = acc[sm][sn][v] + bv;
        if (RES) val += rp[(long)v * N];
        if (BF16OUT) cb[(long)v * N] = f2bf(val);
        else         cp[(long)v * N] = val;
      }
    }
  }
}

// =====================================================================
// Flash attention, one wave per (32-row q-block = 2 q-tiles, head). hd = 64.
// QKV is bf16 (written by the QKV projection GEMM): [S][3072] ushorts,
// Q at cols [0,1024), K at [1024,2048), V at [2048,3072).
// K/V are loaded ONCE per 32 q-rows: 16 WMMAs per 32-key block against
// 4 K-operand b128-load groups and one V stage (2x arithmetic intensity).
// =====================================================================
__global__ void __launch_bounds__(32)
attn_flash_wmma(const unsigned short* __restrict__ qkv,
                float* __restrict__ ctx, int S) {
  __shared__ unsigned short Vlds[32 * 72];  // 32 keys x 64 dims, 16B-aligned rows
  __shared__ unsigned short Plds[32 * 34];  // 2 q-tiles x 16 rows x 32 keys

  const int lane = threadIdx.x;
  const int qt   = blockIdx.x;  // 32-row q block
  const int h    = blockIdx.y;
  const int E3 = 3072, E = 1024;
  const int mlow = lane & 15;
  const int hsel = (lane < 16) ? 0 : 1;

  // ---- Q A-operands for both q-tiles (dims 0..31 / 32..63), resident ----
  V16 qa[2][2];
#pragma unroll
  for (int q = 0; q < 2; ++q) {
    const unsigned short* qp = qkv + (long)(qt * 32 + q * 16 + mlow) * E3 + h * 64;
    const int klow = hsel ? 8 : 0;
    uint4 t;
    t = *(const uint4*)(qp + klow);           qa[q][0].u[0]=t.x; qa[q][0].u[1]=t.y; qa[q][0].u[2]=t.z; qa[q][0].u[3]=t.w;
    t = *(const uint4*)(qp + klow + 16);      qa[q][0].u[4]=t.x; qa[q][0].u[5]=t.y; qa[q][0].u[6]=t.z; qa[q][0].u[7]=t.w;
    t = *(const uint4*)(qp + 32 + klow);      qa[q][1].u[0]=t.x; qa[q][1].u[1]=t.y; qa[q][1].u[2]=t.z; qa[q][1].u[3]=t.w;
    t = *(const uint4*)(qp + 32 + klow + 16); qa[q][1].u[4]=t.x; qa[q][1].u[5]=t.y; qa[q][1].u[6]=t.z; qa[q][1].u[7]=t.w;
  }

  v8f o[2][4];
#pragma unroll
  for (int q = 0; q < 2; ++q)
#pragma unroll
    for (int t = 0; t < 4; ++t) o[q][t] = v8f_zero();
  float m_run[2][8], l_run[2][8];
#pragma unroll
  for (int q = 0; q < 2; ++q)
#pragma unroll
    for (int v = 0; v < 8; ++v) { m_run[q][v] = -3.0e38f; l_run[q][v] = 0.0f; }

  for (int sk = 0; sk < S; sk += 32) {
    // ---- Q·K^T: 2 q-tiles x 2 key sub-tiles, K operands loaded once ----
    v8f st[2][2];
#pragma unroll
    for (int t = 0; t < 2; ++t) {
      const unsigned short* kp = qkv + (long)(sk + t * 16 + mlow) * E3 + E + h * 64;
      const int dlow = hsel ? 8 : 0;
      V16 kb0, kb1;
      uint4 c0 = *(const uint4*)(kp + dlow);
      uint4 c1 = *(const uint4*)(kp + dlow + 16);
      uint4 c2 = *(const uint4*)(kp + 32 + dlow);
      uint4 c3 = *(const uint4*)(kp + 32 + dlow + 16);
      kb0.u[0]=c0.x; kb0.u[1]=c0.y; kb0.u[2]=c0.z; kb0.u[3]=c0.w;
      kb0.u[4]=c1.x; kb0.u[5]=c1.y; kb0.u[6]=c1.z; kb0.u[7]=c1.w;
      kb1.u[0]=c2.x; kb1.u[1]=c2.y; kb1.u[2]=c2.z; kb1.u[3]=c2.w;
      kb1.u[4]=c3.x; kb1.u[5]=c3.y; kb1.u[6]=c3.z; kb1.u[7]=c3.w;
#pragma unroll
      for (int q = 0; q < 2; ++q) {
        v8f z = v8f_zero();
        z = wmma_bf16(qa[q][0], kb0, z);
        z = wmma_bf16(qa[q][1], kb1, z);
        st[q][t] = z;
      }
    }

    // ---- online softmax per q-tile ----
#pragma unroll
    for (int q = 0; q < 2; ++q) {
      float alpha[8];
#pragma unroll
      for (int v = 0; v < 8; ++v) {
        float s0 = st[q][0][v] * 0.125f;  // 1/sqrt(64)
        float s1 = st[q][1][v] * 0.125f;
        float mx = fmaxf(s0, s1);
#pragma unroll
        for (int msk = 1; msk <= 8; msk <<= 1) mx = fmaxf(mx, __shfl_xor(mx, msk, 32));
        float mnew = fmaxf(m_run[q][v], mx);
        float a  = __expf(m_run[q][v] - mnew);
        float p0 = __expf(s0 - mnew);
        float p1 = __expf(s1 - mnew);
        float rs = p0 + p1;
#pragma unroll
        for (int msk = 1; msk <= 8; msk <<= 1) rs += __shfl_xor(rs, msk, 32);
        l_run[q][v] = l_run[q][v] * a + rs;
        m_run[q][v] = mnew;
        alpha[v] = a;
        const int r = q * 16 + v + (hsel ? 8 : 0);
        Plds[r * 34 + mlow]      = f2bf(p0);
        Plds[r * 34 + 16 + mlow] = f2bf(p1);
      }
#pragma unroll
      for (int t = 0; t < 4; ++t)
#pragma unroll
        for (int v = 0; v < 8; ++v) o[q][t][v] *= alpha[v];
    }

    // ---- stage V (32 keys x 64 bf16) -> LDS, one row per lane, b128 copies ----
    {
      const unsigned short* vp = qkv + (long)(sk + lane) * E3 + 2 * E + h * 64;
      unsigned short* d = &Vlds[lane * 72];
#pragma unroll
      for (int j = 0; j < 64; j += 8)
        *(uint4*)(d + j) = *(const uint4*)(vp + j);
    }
    __syncthreads();  // 1-wave WG: S_NOP barrier, acts as compiler LDS fence

    // ---- P A-operands (C-layout -> A-layout via LDS transpose) ----
    V16 pa[2];
#pragma unroll
    for (int q = 0; q < 2; ++q) {
      const unsigned short* p = &Plds[(q * 16 + mlow) * 34];
      const int klow = hsel ? 8 : 0;
#pragma unroll
      for (int j = 0; j < 4; ++j) {
        pa[q].u[j]     = *(const unsigned int*)(p + klow + 2 * j);
        pa[q].u[4 + j] = *(const unsigned int*)(p + klow + 16 + 2 * j);
      }
    }
    // ---- P·V: V operand built once per dim-tile, used by both q-tiles ----
#pragma unroll
    for (int t = 0; t < 4; ++t) {
      V16 vb;
      const int dim = t * 16 + mlow;
      const int k0  = hsel ? 8 : 0;
#pragma unroll
      for (int j = 0; j < 16; ++j) vb.h[j] = Vlds[(k0 + j + (j < 8 ? 0 : 8)) * 72 + dim];
#pragma unroll
      for (int q = 0; q < 2; ++q)
        o[q][t] = wmma_bf16(pa[q], vb, o[q][t]);
    }
    __syncthreads();
  }

  // ---- finalize: O / l, scatter per C layout ----
#pragma unroll
  for (int q = 0; q < 2; ++q)
#pragma unroll
    for (int v = 0; v < 8; ++v) {
      const float inv = 1.0f / l_run[q][v];
      const long row = (long)qt * 32 + q * 16 + v + (hsel ? 8 : 0);
#pragma unroll
      for (int t = 0; t < 4; ++t) {
        const long col = h * 64 + t * 16 + mlow;
        ctx[row * 1024 + col] = o[q][t][v] * inv;
      }
    }
}

// =====================================================================
// Linear upsample L -> T (T = 2L here), half-pixel centers, clamped.
// =====================================================================
__global__ void upsample_linear(const float* __restrict__ src,
                                float* __restrict__ dst, int L, int T) {
  int idx = blockIdx.x * blockDim.x + threadIdx.x;
  int total = T * 1024;
  if (idx >= total) return;
  int i = idx >> 10, e = idx & 1023;
  float s = (i + 0.5f) * ((float)L / (float)T) - 0.5f;
  s = fminf(fmaxf(s, 0.0f), (float)(L - 1));
  int i0 = (int)floorf(s);
  int i1 = min(i0 + 1, L - 1);
  float w = s - (float)i0;
  dst[idx] = src[(long)i0 * 1024 + e] * (1.0f - w) + src[(long)i1 * 1024 + e] * w;
}

// =====================================================================
extern "C" void kernel_launch(void* const* d_in, const int* in_sizes, int n_in,
                              void* d_out, int out_size, void* d_ws, size_t ws_size,
                              hipStream_t stream) {
  const float* query = (const float*)d_in[0];
  const float* in_w  = (const float*)d_in[1];
  const float* in_b  = (const float*)d_in[2];
  const float* ow    = (const float*)d_in[3];
  const float* ob    = (const float*)d_in[4];
  const float* uw    = (const float*)d_in[5];
  const float* ub    = (const float*)d_in[6];
  float* out = (float*)d_out;

  const int E = 1024, E3 = 3072, S0 = 2048;

  // workspace layout
  unsigned short* QKVb = (unsigned short*)d_ws;             // 3*2048*1024 bf16
  float* fbase = (float*)((char*)d_ws + (size_t)3 * S0 * E * sizeof(unsigned short));
  float* CTX  = fbase;                                      // 2048*1024 fp32
  float* OUTS = CTX + (size_t)S0 * E;                       // 3,932,160 fp32
  const size_t outs_off[4] = {0,
                              (size_t)2048 * 1024,
                              (size_t)2048 * 1024 + (size_t)1024 * 1024,
                              (size_t)2048 * 1024 + (size_t)1024 * 1024 + (size_t)512 * 1024};
  float* CUR  = OUTS + (size_t)(2048 + 1024 + 512 + 256) * 1024;  // 1024*1024
  float* UPS  = CUR + (size_t)1024 * 1024;                         // 2048*1024

  // ---- per-level MHA ----
  for (int lvl = 0; lvl < 4; ++lvl) {
    const int S = S0 >> lvl;
    const long lda = (long)(1 << lvl) * E;  // strided view query[:, ::stride, :]

    dim3 g1(S / BM, E3 / BN);
    gemm_bias_wmma<false, true><<<g1, 256, 0, stream>>>(
        query, lda, in_w + (size_t)lvl * E3 * E, (long)E,
        in_b + (size_t)lvl * E3, nullptr, nullptr, QKVb, S, E3, E);

    dim3 g2(S / 32, 16);
    attn_flash_wmma<<<g2, 32, 0, stream>>>(QKVb, CTX, S);

    dim3 g3(S / BM, E / BN);
    gemm_bias_wmma<false, false><<<g3, 256, 0, stream>>>(
        CTX, (long)E, ow + (size_t)lvl * E * E, (long)E,
        ob + (size_t)lvl * E, nullptr, OUTS + outs_off[lvl], nullptr, S, E, E);
  }

  // ---- hierarchical fuse: upsample -> up-proj (+ fine residual) ----
  const float* cur = OUTS + outs_off[3];  // level-3 output, length 256
  for (int lvl = 2; lvl >= 0; --lvl) {
    const int Lc = S0 >> (lvl + 1);
    const int Lt = S0 >> lvl;
    const int n = Lt * E;
    upsample_linear<<<(n + 255) / 256, 256, 0, stream>>>(cur, UPS, Lc, Lt);

    float* dst = (lvl == 0) ? out : CUR;
    dim3 g(Lt / BM, E / BN);
    gemm_bias_wmma<true, false><<<g, 256, 0, stream>>>(
        UPS, (long)E, uw + (size_t)lvl * E * E, (long)E,
        ub + (size_t)lvl * E, OUTS + outs_off[lvl], dst, nullptr, Lt, E, E);
    cur = CUR;
  }
}
